// GCN_C_20529943675404
// MI455X (gfx1250) — compile-verified, matrix operand order checked by reading, not run on hardware
//
#include <hip/hip_runtime.h>

#define Nn   12288
#define FIN  500
#define HID  64
#define OUTC 16

typedef __attribute__((ext_vector_type(16))) __bf16 v16bf;
typedef __attribute__((ext_vector_type(8)))  float  v8f;
typedef __attribute__((ext_vector_type(4)))  float  v4f;

// Load 16 consecutive f32 (non-temporal: adj is streamed once per pass,
// 604MB >> 192MB L2, so don't pollute the cache) and convert to bf16 packed
// ascending -> contiguous k = 16*hi + s per lane, the order this file uses for both A and B.
__device__ __forceinline__ v16bf cvt16_nt(const float* __restrict__ p) {
  v16bf r;
  const v4f* q = (const v4f*)p;
#pragma unroll
  for (int i = 0; i < 4; ++i) {
    v4f v = __builtin_nontemporal_load(q + i);
    r[4*i+0] = (__bf16)v.x;
    r[4*i+1] = (__bf16)v.y;
    r[4*i+2] = (__bf16)v.z;
    r[4*i+3] = (__bf16)v.w;
  }
  return r;
}

// ---------------------------------------------------------------------------
// Kernel A: H1^T = (x @ W1 + b1)^T  stored bf16, shape [HID][Nn].
// Thread = (node, feature-group-of-8): x value loaded once per 8 FMAs,
// stores are contiguous along nodes (coalesced).
// ---------------------------------------------------------------------------
__global__ void __launch_bounds__(256)
gcn_lin1(const float* __restrict__ x, const float* __restrict__ W1,
         const float* __restrict__ b1, __bf16* __restrict__ h1t) {
  const int node = blockIdx.x * blockDim.x + threadIdx.x;
  const int fg   = blockIdx.y;                       // 8 features per thread
  const float* xr = x + (size_t)node * FIN;
  float acc[8];
#pragma unroll
  for (int j = 0; j < 8; ++j) acc[j] = b1[fg * 8 + j];
  for (int k = 0; k < FIN; ++k) {
    const float xv = xr[k];
    const float* wrow = W1 + (size_t)k * HID + fg * 8;
#pragma unroll
    for (int j = 0; j < 8; ++j) acc[j] = fmaf(xv, wrow[j], acc[j]);
  }
#pragma unroll
  for (int j = 0; j < 8; ++j)
    h1t[(size_t)(fg * 8 + j) * Nn + node] = (__bf16)acc[j];
}

// ---------------------------------------------------------------------------
// Kernel B: Z = adj @ H1 (WMMA bf16, f32 accum), fused epilogue:
// H2 = relu(Z) @ W2 + b2, stored transposed bf16 [OUTC][Nn].
// One wave owns a 16-row tile of adj and all 64 output columns
// (4 accumulators) so each adj row is read exactly once.
// ---------------------------------------------------------------------------
__global__ void __launch_bounds__(128)
gcn_agg1(const float* __restrict__ adj, const __bf16* __restrict__ h1t,
         const float* __restrict__ W2, const float* __restrict__ b2,
         __bf16* __restrict__ h2t) {
  __shared__ float lds[4][16 * 64];                  // per-wave 16x64 relu(Z)

  const int lane = threadIdx.x & 31;
  const int wave = threadIdx.x >> 5;
  const int l16  = lane & 15;
  const int hi   = lane >> 4;                        // K-half / M-half select
  const int m0   = (blockIdx.x * 4 + wave) * 16;

  v8f acc0 = {}, acc1 = {}, acc2 = {}, acc3 = {};

  const float*  arow = adj + (size_t)(m0 + l16) * Nn + hi * 16;
  const __bf16* brow = h1t + (size_t)l16 * Nn + hi * 16;   // B rows = features

  for (int kc = 0; kc < Nn; kc += 32) {
    v16bf a  = cvt16_nt(arow + kc);
    v16bf bA = *(const v16bf*)(brow + kc);                 // cols  0..15
    v16bf bB = *(const v16bf*)(brow + (size_t)16 * Nn + kc); // 16..31
    v16bf bC = *(const v16bf*)(brow + (size_t)32 * Nn + kc); // 32..47
    v16bf bD = *(const v16bf*)(brow + (size_t)48 * Nn + kc); // 48..63
    acc0 = __builtin_amdgcn_wmma_f32_16x16x32_bf16(false, a, false, bA, (short)0, acc0, false, false);
    acc1 = __builtin_amdgcn_wmma_f32_16x16x32_bf16(false, a, false, bB, (short)0, acc1, false, false);
    acc2 = __builtin_amdgcn_wmma_f32_16x16x32_bf16(false, a, false, bC, (short)0, acc2, false, false);
    acc3 = __builtin_amdgcn_wmma_f32_16x16x32_bf16(false, a, false, bD, (short)0, acc3, false, false);
  }

  // Spill relu(Z) tile to LDS (C/D layout: VGPR r, lane -> M = r + 8*hi, N = l16 + 16c)
  float* my = lds[wave];
#pragma unroll
  for (int r = 0; r < 8; ++r) {
    const int m = r + 8 * hi;
    my[m * 64 + l16 +  0] = fmaxf(acc0[r], 0.0f);
    my[m * 64 + l16 + 16] = fmaxf(acc1[r], 0.0f);
    my[m * 64 + l16 + 32] = fmaxf(acc2[r], 0.0f);
    my[m * 64 + l16 + 48] = fmaxf(acc3[r], 0.0f);
  }
  __syncthreads();

  // Fused tiny GEMM: H2[m][n] = sum_k relu(Z)[m][k]*W2[k][n] + b2[n]
  // lane -> row m = l16, 8 output cols; store transposed bf16.
  const int m = l16;
#pragma unroll
  for (int j = 0; j < 8; ++j) {
    const int n = hi * 8 + j;
    float s = b2[n];
#pragma unroll 8
    for (int kk = 0; kk < HID; ++kk)
      s = fmaf(my[m * 64 + kk], W2[kk * OUTC + n], s);
    h2t[(size_t)n * Nn + (m0 + m)] = (__bf16)s;
  }
}

// ---------------------------------------------------------------------------
// Kernel C: out = adj @ H2   (12288x16, f32 output)
// ---------------------------------------------------------------------------
__global__ void __launch_bounds__(128)
gcn_agg2(const float* __restrict__ adj, const __bf16* __restrict__ h2t,
         float* __restrict__ out) {
  const int lane = threadIdx.x & 31;
  const int wave = threadIdx.x >> 5;
  const int l16  = lane & 15;
  const int hi   = lane >> 4;
  const int m0   = (blockIdx.x * 4 + wave) * 16;

  v8f acc = {};
  const float*  arow = adj + (size_t)(m0 + l16) * Nn + hi * 16;
  const __bf16* brow = h2t + (size_t)l16 * Nn + hi * 16;

  for (int kc = 0; kc < Nn; kc += 32) {
    v16bf a = cvt16_nt(arow + kc);
    v16bf b = *(const v16bf*)(brow + kc);
    acc = __builtin_amdgcn_wmma_f32_16x16x32_bf16(false, a, false, b, (short)0, acc, false, false);
  }

#pragma unroll
  for (int r = 0; r < 8; ++r)
    out[(size_t)(m0 + r + 8 * hi) * OUTC + l16] = acc[r];
}

// ---------------------------------------------------------------------------
extern "C" void kernel_launch(void* const* d_in, const int* in_sizes, int n_in,
                              void* d_out, int out_size, void* d_ws, size_t ws_size,
                              hipStream_t stream) {
  const float* x   = (const float*)d_in[0];
  const float* adj = (const float*)d_in[1];
  const float* W1  = (const float*)d_in[2];
  const float* b1  = (const float*)d_in[3];
  const float* W2  = (const float*)d_in[4];
  const float* b2  = (const float*)d_in[5];
  float* out = (float*)d_out;

  __bf16* h1t = (__bf16*)d_ws;                                   // HID  x Nn bf16 (1.5 MB)
  __bf16* h2t = (__bf16*)((char*)d_ws +
                          (size_t)HID * Nn * sizeof(__bf16));    // OUTC x Nn bf16 (0.4 MB)

  dim3 gridA(Nn / 256, HID / 8);
  gcn_lin1<<<gridA, 256, 0, stream>>>(x, W1, b1, h1t);

  const int tiles  = Nn / 16;          // 768 row tiles, 4 waves/block
  const int blocks = tiles / 4;        // 192
  gcn_agg1<<<blocks, 128, 0, stream>>>(adj, h1t, W2, b2, h2t);
  gcn_agg2<<<blocks, 128, 0, stream>>>(adj, h2t, out);
}
